// TransformerM_7541962572068
// MI455X (gfx1250) — matrix operand order repaired and checked by reading, not compiled, not run on hardware
//
#include <hip/hip_runtime.h>
#include <math.h>

// ---------------- problem dims ----------------
#define NB   64
#define SS   200
#define TT   200
#define AA   40
#define DD   512
#define HH   8
#define DKK  64
#define DVV  64
#define DFFF 2048
#define VV   100
#define LL   6
#define CH   8            // batch chunk for gbf feature GEMMs

// ---------------- types ----------------
typedef __bf16 bf16;
typedef __attribute__((ext_vector_type(16))) bf16  v16bf;
typedef __attribute__((ext_vector_type(8)))  float v8f;

// native RNE convert (backend emits v_cvt_*_bf16 ops instead of 3-op bit math)
__device__ __forceinline__ bf16 f2bf(float f) { return (bf16)f; }

__device__ __forceinline__ v8f v8f_zero() {
    v8f z;
#pragma unroll
    for (int i = 0; i < 8; ++i) z[i] = 0.f;
    return z;
}

// =================================================================
// Tiled WMMA GEMM:  C = act(alpha*A@B + bias) + Res
// A: M x K (fp32, or bf16 already in global), B: K x Nc fp32 (opt. transposed),
// C: M x Nc (fp32 or bf16).  Block 256 thr = 8 waves, tile 128x128,
// wave tile 32x64 (2x4 of 16x16), K-step 32, bf16 operands staged in LDS.
// ABF16 path stages the A tile with global_load_async_to_lds_b128
// (requires M % 128 == 0, guaranteed at call sites).
// =================================================================
template <int ACT, bool ABF16, bool BT, bool OBF16>
__global__ __launch_bounds__(256) void gemm_wmma(
    const void* __restrict__ Aptr, const float* __restrict__ B,
    const float* __restrict__ bias, const float* __restrict__ Res,
    void* __restrict__ Cptr, int M, int K, int Nc,
    int lda, int ldb, int ldc, float alpha)
{
    __shared__ bf16 As[128 * 32];
    __shared__ bf16 Bs[32 * 128];

    const int t    = threadIdx.x;
    const int lane = t & 31;
    const int wave = t >> 5;
    const int wm   = wave >> 1;   // 0..3  (rows 32*wm)
    const int wn   = wave & 1;    // 0..1  (cols 64*wn)
    const int r0   = blockIdx.x * 128;
    const int c0   = blockIdx.y * 128;
    const int colL = lane & 15;
    const int rg   = lane >> 4;

    v8f acc[2][4];
#pragma unroll
    for (int i = 0; i < 2; ++i)
#pragma unroll
        for (int j = 0; j < 4; ++j) acc[i][j] = v8f_zero();

    const unsigned lds_As = (unsigned)(uintptr_t)(void*)As;

    for (int kt = 0; kt < K; kt += 32) {
        // ---- stage A tile 128x32 -> LDS (bf16) ----
        if (ABF16) {
            // async DMA: 512 x b128 segments, 2 per thread; tracked on ASYNCcnt
#pragma unroll
            for (int i = 0; i < 2; ++i) {
                int seg  = t * 2 + i;
                int row  = seg >> 2;
                int part = seg & 3;
                unsigned goff  = (unsigned)(((size_t)(r0 + row) * lda + kt + part * 8) * 2);
                unsigned laddr = lds_As + (unsigned)((row * 32 + part * 8) * 2);
                asm volatile("global_load_async_to_lds_b128 %0, %1, %2"
                             :: "v"(laddr), "v"(goff), "s"((const void*)Aptr)
                             : "memory");
            }
        } else {
            int row = t >> 1;
            int cb  = (t & 1) * 16;
            int gr  = r0 + row;
            const float* Af = (const float*)Aptr;
#pragma unroll
            for (int i = 0; i < 16; ++i) {
                float v = 0.f;
                if (gr < M) v = Af[(size_t)gr * lda + kt + cb + i];
                As[row * 32 + cb + i] = f2bf(v);
            }
        }
        // ---- stage B tile 32x128 -> LDS (fp32 -> bf16 via native cvt) ----
        {
            int row = t >> 3;            // 0..31
            int cb  = (t & 7) * 16;      // 0..112
#pragma unroll
            for (int i = 0; i < 16; ++i) {
                int col = c0 + cb + i;
                float v = 0.f;
                if (col < Nc)
                    v = BT ? B[(size_t)col * ldb + kt + row]
                           : B[(size_t)(kt + row) * ldb + col];
                Bs[row * 128 + cb + i] = f2bf(v);
            }
        }
        if (ABF16)
            asm volatile("s_wait_asynccnt 0" ::: "memory");
        __syncthreads();

        // ---- fragments + 8 WMMAs per wave ----
        v16bf afr[2], bfr[4];
#pragma unroll
        for (int tm = 0; tm < 2; ++tm) {
            int arow = wm * 32 + tm * 16 + colL;
#pragma unroll
            for (int e = 0; e < 16; ++e) {
                int kk = e + rg * 16;
                afr[tm][e] = As[arow * 32 + kk];
            }
        }
#pragma unroll
        for (int tn = 0; tn < 4; ++tn) {
            int bcol = wn * 64 + tn * 16 + colL;
#pragma unroll
            for (int e = 0; e < 16; ++e) {
                int kk = e + rg * 16;
                bfr[tn][e] = Bs[kk * 128 + bcol];
            }
        }
#pragma unroll
        for (int tm = 0; tm < 2; ++tm)
#pragma unroll
            for (int tn = 0; tn < 4; ++tn)
                acc[tm][tn] = __builtin_amdgcn_wmma_f32_16x16x32_bf16(
                    false, afr[tm], false, bfr[tn], (short)0, acc[tm][tn],
                    false, false);
        __syncthreads();
    }

    // ---- epilogue ----
#pragma unroll
    for (int tm = 0; tm < 2; ++tm)
#pragma unroll
        for (int tn = 0; tn < 4; ++tn)
#pragma unroll
            for (int r = 0; r < 8; ++r) {
                int row = r0 + wm * 32 + tm * 16 + rg * 8 + r;
                int col = c0 + wn * 64 + tn * 16 + colL;
                if (row < M && col < Nc) {
                    float v = acc[tm][tn][r] * alpha;
                    if (bias) v += bias[col];
                    if (ACT == 1) v = fmaxf(v, 0.f);
                    else if (ACT == 2) v = 0.5f * v * (1.f + erff(v * 0.70710678118f));
                    if (Res) v += Res[(size_t)row * ldc + col];
                    if (OBF16)
                        ((bf16*)Cptr)[(size_t)row * ldc + col] = f2bf(v);
                    else
                        ((float*)Cptr)[(size_t)row * ldc + col] = v;
                }
            }
}

// =================================================================
// Flash attention (wave32).  Block = 64 thr = 2 waves, each wave owns
// 16 query rows; keys processed in 32-wide tiles; online softmax.
// mode 0: encoder self (src pad mask + gaussian bias via inv/ab)
// mode 1: decoder self (causal + trg pad mask)
// mode 2: cross (src pad mask)
// Q/O: (N, Lq, H*DK) fp32.  K/V: (N, Lk, H*DK) fp32.
// =================================================================
__global__ __launch_bounds__(64) void flash_attn(
    const float* __restrict__ Q, const float* __restrict__ Km,
    const float* __restrict__ Vm, float* __restrict__ O,
    const int* __restrict__ kseq, const int* __restrict__ inv,
    const float* __restrict__ ab, int Lq, int Lk, int mode)
{
    const int lane = threadIdx.x & 31;
    const int wave = threadIdx.x >> 5;
    const int nh   = blockIdx.x;
    const int n    = nh / HH;
    const int h    = nh % HH;
    const int qbase = blockIdx.y * 32 + wave * 16;
    const int colL = lane & 15;
    const int rg   = lane >> 4;
    const int ld   = HH * DKK;

    __shared__ bf16 Plds[2][16 * 32];

    const float* Qb = Q  + (size_t)n * Lq * ld + h * DKK;
    const float* Kb = Km + (size_t)n * Lk * ld + h * DKK;
    const float* Vb = Vm + (size_t)n * Lk * ld + h * DVV;

    // Q fragments (A layout), two dk-halves of 32
    v16bf aq[2];
    {
        int qr = qbase + colL;
        int qc = qr < Lq ? qr : (Lq - 1);
#pragma unroll
        for (int half = 0; half < 2; ++half)
#pragma unroll
            for (int e = 0; e < 16; ++e) {
                int kk = e + rg * 16 + half * 32;
                aq[half][e] = f2bf(Qb[(size_t)qc * ld + kk]);
            }
    }

    float mrow[8], lrow[8];
    int   qg[8], invq[8];
    v8f   o[4];
#pragma unroll
    for (int r = 0; r < 8; ++r) {
        mrow[r] = -1e30f; lrow[r] = 0.f;
        qg[r]   = qbase + rg * 8 + r;
        invq[r] = (mode == 0 && qg[r] < Lq) ? inv[n * SS + qg[r]] : -1;
    }
#pragma unroll
    for (int t4 = 0; t4 < 4; ++t4) o[t4] = v8f_zero();

    const float scale = 0.125f;  // 1/sqrt(DK)

    for (int kt = 0; kt < Lk; kt += 32) {
        // ---- scores: two 16x16 tiles over 32 keys ----
        v8f s[2];
#pragma unroll
        for (int sub = 0; sub < 2; ++sub) {
            int key  = kt + sub * 16 + colL;
            int keyc = key < Lk ? key : (Lk - 1);
            v16bf b0, b1;
#pragma unroll
            for (int e = 0; e < 16; ++e) {
                int dk0 = rg * 16 + e;
                b0[e] = f2bf(Kb[(size_t)keyc * ld + dk0]);
                b1[e] = f2bf(Kb[(size_t)keyc * ld + 32 + dk0]);
            }
            v8f z = v8f_zero();
            s[sub] = __builtin_amdgcn_wmma_f32_16x16x32_bf16(
                false, aq[0], false, b0, (short)0, z, false, false);
            s[sub] = __builtin_amdgcn_wmma_f32_16x16x32_bf16(
                false, aq[1], false, b1, (short)0, s[sub], false, false);
        }

        // ---- scale + bias + mask ----
#pragma unroll
        for (int sub = 0; sub < 2; ++sub) {
            int  key  = kt + sub * 16 + colL;
            bool kval = key < Lk;
            int  tok  = kval ? kseq[n * Lk + key] : 0;
            int  invk = (mode == 0 && kval) ? inv[n * Lk + key] : -1;
#pragma unroll
            for (int r = 0; r < 8; ++r) {
                float v = s[sub][r] * scale;
                if (mode == 0 && invq[r] >= 0 && invk >= 0)
                    v += ab[(((size_t)n * AA + invq[r]) * AA + invk) * HH + h];
                bool ok = kval && (tok != 0);
                if (mode == 1) ok = ok && (key <= qg[r]);
                s[sub][r] = ok ? v : -1e9f;
            }
        }

        // ---- online softmax ----
        float corr[8];
#pragma unroll
        for (int r = 0; r < 8; ++r) {
            float mx = fmaxf(s[0][r], s[1][r]);
            for (int off = 1; off < 16; off <<= 1)
                mx = fmaxf(mx, __shfl_xor(mx, off, 32));
            float mnew = fmaxf(mrow[r], mx);
            corr[r] = expf(mrow[r] - mnew);
            float p0 = expf(s[0][r] - mnew);
            float p1 = expf(s[1][r] - mnew);
            s[0][r] = p0; s[1][r] = p1;
            float ps = p0 + p1;
            for (int off = 1; off < 16; off <<= 1)
                ps += __shfl_xor(ps, off, 32);
            lrow[r] = lrow[r] * corr[r] + ps;
            mrow[r] = mnew;
        }
#pragma unroll
        for (int t4 = 0; t4 < 4; ++t4)
#pragma unroll
            for (int r = 0; r < 8; ++r) o[t4][r] *= corr[r];

        // ---- P (C layout) -> LDS -> A layout ----
#pragma unroll
        for (int sub = 0; sub < 2; ++sub)
#pragma unroll
            for (int r = 0; r < 8; ++r)
                Plds[wave][(rg * 8 + r) * 32 + sub * 16 + colL] = f2bf(s[sub][r]);
        __syncthreads();
        v16bf ap;
#pragma unroll
        for (int e = 0; e < 16; ++e) {
            int kk = e + rg * 16;
            ap[e] = Plds[wave][colL * 32 + kk];
        }
        __syncthreads();

        // ---- O += P @ V ----
#pragma unroll
        for (int t4 = 0; t4 < 4; ++t4) {
            v16bf bv;
#pragma unroll
            for (int e = 0; e < 16; ++e) {
                int key  = kt + rg * 16 + e;
                int keyc = key < Lk ? key : (Lk - 1);
                bv[e] = f2bf(Vb[(size_t)keyc * ld + t4 * 16 + colL]);
            }
            o[t4] = __builtin_amdgcn_wmma_f32_16x16x32_bf16(
                false, ap, false, bv, (short)0, o[t4], false, false);
        }
    }

    // ---- finalize ----
#pragma unroll
    for (int r = 0; r < 8; ++r) {
        float li   = lrow[r];
        float linv = li > 0.f ? 1.f / li : 0.f;
        int   q    = qg[r];
        if (q < Lq) {
#pragma unroll
            for (int t4 = 0; t4 < 4; ++t4)
                O[((size_t)n * Lq + q) * ld + h * DVV + t4 * 16 + colL] =
                    o[t4][r] * linv;
        }
    }
}

// ================= elementwise / feature kernels =================

__global__ __launch_bounds__(256) void embed_kernel(
    const int* __restrict__ seq, const float* __restrict__ E,
    float* __restrict__ x, const float* __restrict__ scale_ptr, int Lseq)
{
    size_t i = (size_t)blockIdx.x * 256 + threadIdx.x;
    size_t tot = (size_t)NB * Lseq * DD;
    if (i >= tot) return;
    int    d  = (int)(i % DD);
    size_t ns = i / DD;
    int    s  = (int)(ns % Lseq);
    int  tok  = seq[ns];
    float den = powf(10000.f, (2.f * (d >> 1)) / (float)DD);
    float ang = (float)s / den;
    float pv  = (d & 1) ? cosf(ang) : sinf(ang);
    float sc  = scale_ptr ? scale_ptr[0] : 1.f;
    x[i] = sc * (E[(size_t)tok * DD + d] + pv);
}

__global__ __launch_bounds__(256) void ln_kernel(
    float* __restrict__ x, const float* __restrict__ g,
    const float* __restrict__ b)
{
    __shared__ float s1[256], s2[256];
    const int t = threadIdx.x;
    float* row = x + (size_t)blockIdx.x * DD;
    float v0 = row[t], v1 = row[t + 256];
    s1[t] = v0 + v1;
    s2[t] = v0 * v0 + v1 * v1;
    __syncthreads();
    for (int o = 128; o > 0; o >>= 1) {
        if (t < o) { s1[t] += s1[t + o]; s2[t] += s2[t + o]; }
        __syncthreads();
    }
    float mean = s1[0] * (1.f / DD);
    float var  = s2[0] * (1.f / DD) - mean * mean;
    float rstd = rsqrtf(var + 1e-6f);
    row[t]       = (v0 - mean) * rstd * g[t] + b[t];
    row[t + 256] = (v1 - mean) * rstd * g[t + 256] + b[t + 256];
}

__global__ __launch_bounds__(256) void fill_inv(int* __restrict__ inv) {
    size_t i = (size_t)blockIdx.x * 256 + threadIdx.x;
    if (i < (size_t)NB * SS) inv[i] = -1;
}
__global__ __launch_bounds__(256) void scat_inv(
    const int* __restrict__ atoms, int* __restrict__ inv) {
    size_t i = (size_t)blockIdx.x * 256 + threadIdx.x;
    if (i >= (size_t)NB * AA) return;
    int n = (int)(i / AA);
    inv[(size_t)n * SS + atoms[i]] = (int)(i % AA);
}

// ef (bf16) for a batch chunk: block per (n_local, a)
__global__ __launch_bounds__(256) void ef_kernel(
    const float* __restrict__ pos, int n0, bf16* __restrict__ ef,
    const float* __restrict__ means, const float* __restrict__ stds,
    const float* __restrict__ mulp, const float* __restrict__ biasp)
{
    __shared__ float distS[AA];
    __shared__ int   maskS[AA];
    const int t  = threadIdx.x;
    const int nl = blockIdx.x / AA;
    const int a  = blockIdx.x % AA;
    const int n  = n0 + nl;
    if (t < AA) {
        const float* pa = pos + ((size_t)n * AA + a) * 3;
        const float* pb = pos + ((size_t)n * AA + t) * 3;
        float dx = pa[0] - pb[0], dy = pa[1] - pb[1], dz = pa[2] - pb[2];
        distS[t] = sqrtf(dx * dx + dy * dy + dz * dz);
        maskS[t] = (pb[0] == 0.f && pb[1] == 0.f && pb[2] == 0.f);
    }
    __syncthreads();
    const float mul = mulp[0], bia = biasp[0];
    for (int b = 0; b < AA; ++b) {
        float  xg   = mul * distS[b] + bia;
        int    msk  = maskS[b];
        size_t base = (((size_t)nl * AA + a) * AA + b) * DD;
        for (int d = t; d < DD; d += 256) {
            float sd  = fabsf(stds[d]) + 1e-5f;
            float ivs = 1.f / sd;
            float z   = (xg - means[d]) * ivs;
            float v   = msk ? 0.f : expf(-0.5f * z * z) * 0.3989422804014327f * ivs;
            ef[base + d] = f2bf(v);
        }
    }
}

// epos[n,a,d] = sum_b ef[n,a,b,d] (fp32, recomputed): block per (n,a)
__global__ __launch_bounds__(256) void epos_kernel(
    const float* __restrict__ pos, float* __restrict__ epos,
    const float* __restrict__ means, const float* __restrict__ stds,
    const float* __restrict__ mulp, const float* __restrict__ biasp)
{
    __shared__ float distS[AA];
    __shared__ int   maskS[AA];
    const int t  = threadIdx.x;
    const int na = blockIdx.x;             // n*AA + a
    const int n  = na / AA;
    const int a  = na % AA;
    if (t < AA) {
        const float* pa = pos + ((size_t)n * AA + a) * 3;
        const float* pb = pos + ((size_t)n * AA + t) * 3;
        float dx = pa[0] - pb[0], dy = pa[1] - pb[1], dz = pa[2] - pb[2];
        distS[t] = sqrtf(dx * dx + dy * dy + dz * dz);
        maskS[t] = (pb[0] == 0.f && pb[1] == 0.f && pb[2] == 0.f);
    }
    __syncthreads();
    const float mul = mulp[0], bia = biasp[0];
    for (int d = t; d < DD; d += 256) {
        float sd  = fabsf(stds[d]) + 1e-5f;
        float ivs = 1.f / sd;
        float c   = 0.3989422804014327f * ivs;
        float acc = 0.f;
        for (int b = 0; b < AA; ++b)
            if (!maskS[b]) {
                float z = (mul * distS[b] + bia - means[d]) * ivs;
                acc += expf(-0.5f * z * z) * c;
            }
        epos[(size_t)na * DD + d] = acc;
    }
}

// x[n, atoms[n,j], :] += rate2 * epos[n,j,:]
__global__ __launch_bounds__(256) void scatter_epos(
    const int* __restrict__ atoms, const float* __restrict__ epos,
    const float* __restrict__ rate2, float* __restrict__ x)
{
    size_t i = (size_t)blockIdx.x * 256 + threadIdx.x;
    if (i >= (size_t)NB * AA * DD) return;
    int    d  = (int)(i % DD);
    size_t nj = i / DD;
    int    n  = (int)(nj / AA);
    int    s  = atoms[nj];
    x[((size_t)n * SS + s) * DD + d] += rate2[0] * epos[i];
}

// ================= host side =================

// param input indices (insertion-order flattening of setup_inputs dict)
enum {
    IN_SRC_SEQ = 0, IN_TRG_SEQ, IN_SRC_POS, IN_SRC_Z, IN_SRC_INDEX, IN_SRC_ATOMS,
    P_EMB, P_RATE1, P_RATE2,
    P_GBF_MEANS, P_GBF_STDS, P_GBF_MUL, P_GBF_BIAS,
    P_GBFP_W1, P_GBFP_B1, P_GBFP_W2, P_GBFP_B2,
    P_ENC_LN_G, P_ENC_LN_B, P_DEC_LN_G, P_DEC_LN_B,
    P_ENC_BASE   // enc layer l: +l*12 {wq,wk,wv,wo,ln_g,ln_b,w1,b1,w2,b2,ln_g,ln_b}
};
// dec layer l base: P_ENC_BASE + 72 + l*18 {self x6, cross x6, ffn x6}

extern "C" void kernel_launch(void* const* d_in, const int* in_sizes, int n_in,
                              void* d_out, int out_size, void* d_ws, size_t ws_size,
                              hipStream_t stream)
{
    (void)in_sizes; (void)n_in; (void)out_size; (void)ws_size;
    auto F  = [&](int i) { return (const float*)d_in[i]; };
    auto Ii = [&](int i) { return (const int*)d_in[i]; };

    const size_t NS = (size_t)NB * SS;   // 12800 rows

    // ---- workspace arena ----
    char*  W   = (char*)d_ws;
    size_t off = 0;
    auto alloc = [&](size_t bytes) {
        size_t o = off;
        off += (bytes + 255) & ~(size_t)255;
        return o;
    };
    float* x  = (float*)(W + alloc(NS * DD * 4));
    float* y  = (float*)(W + alloc(NS * DD * 4));
    float* qb = (float*)(W + alloc(NS * DD * 4));
    float* kb = (float*)(W + alloc(NS * DD * 4));
    float* vb = (float*)(W + alloc(NS * DD * 4));
    float* ob = (float*)(W + alloc(NS * DD * 4));
    char*  hreg = W + alloc(NS * DFFF * 2);                 // 52.4 MB, reused
    bf16*  hffn = (bf16*)hreg;                              // FFN hidden (bf16)
    bf16*  efc  = (bf16*)hreg;                              // ef chunk (bf16)
    bf16*  h1c  = (bf16*)(hreg + (size_t)CH * AA * AA * DD * 2);
    float* ab   = (float*)(W + alloc((size_t)NB * AA * AA * HH * 4));
    float* epos = (float*)(W + alloc((size_t)NB * AA * DD * 4));
    int*   inv  = (int*)(W + alloc((size_t)NB * SS * 4));

    // ---- GEMM launch helpers ----
    auto g_plain = [&](const float* A, const float* B, const float* bias,
                       const float* res, float* C, int M, int K, int Nc) {
        dim3 g((M + 127) / 128, (Nc + 127) / 128);
        gemm_wmma<0, false, false, false><<<g, 256, 0, stream>>>(
            A, B, bias, res, C, M, K, Nc, K, Nc, Nc, 1.f);
    };
    auto g_relu_bf = [&](const float* A, const float* B, const float* bias,
                         bf16* C, int M, int K, int Nc) {
        dim3 g((M + 127) / 128, (Nc + 127) / 128);
        gemm_wmma<1, false, false, true><<<g, 256, 0, stream>>>(
            A, B, bias, nullptr, C, M, K, Nc, K, Nc, Nc, 1.f);
    };
    auto g_abf = [&](const bf16* A, const float* B, const float* bias,
                     const float* res, float* C, int M, int K, int Nc) {
        dim3 g((M + 127) / 128, (Nc + 127) / 128);
        gemm_wmma<0, true, false, false><<<g, 256, 0, stream>>>(
            A, B, bias, res, C, M, K, Nc, K, Nc, Nc, 1.f);
    };

    // ================= encoder front end =================
    embed_kernel<<<(unsigned)((NS * DD + 255) / 256), 256, 0, stream>>>(
        Ii(IN_SRC_SEQ), F(P_EMB), x, F(P_RATE1), SS);

    fill_inv<<<(unsigned)((NB * SS + 255) / 256), 256, 0, stream>>>(inv);
    scat_inv<<<(unsigned)((NB * AA + 255) / 256), 256, 0, stream>>>(
        Ii(IN_SRC_ATOMS), inv);

    epos_kernel<<<NB * AA, 256, 0, stream>>>(
        F(IN_SRC_POS), epos, F(P_GBF_MEANS), F(P_GBF_STDS),
        F(P_GBF_MUL), F(P_GBF_BIAS));
    scatter_epos<<<(unsigned)(((size_t)NB * AA * DD + 255) / 256), 256, 0, stream>>>(
        Ii(IN_SRC_ATOMS), epos, F(P_RATE2), x);
    ln_kernel<<<(unsigned)NS, 256, 0, stream>>>(x, F(P_ENC_LN_G), F(P_ENC_LN_B));

    // gaussian bias projection, chunked over batch: ef -> gelu(ef@w1+b1) -> @w2+b2
    for (int c = 0; c < NB / CH; ++c) {
        ef_kernel<<<CH * AA, 256, 0, stream>>>(
            F(IN_SRC_POS), c * CH, efc, F(P_GBF_MEANS), F(P_GBF_STDS),
            F(P_GBF_MUL), F(P_GBF_BIAS));
        const int Mg = CH * AA * AA;   // 12800 (multiple of 128 for async A path)
        dim3 g1((Mg + 127) / 128, (DD + 127) / 128);
        gemm_wmma<2, true, false, true><<<g1, 256, 0, stream>>>(
            efc, F(P_GBFP_W1), F(P_GBFP_B1), nullptr, h1c,
            Mg, DD, DD, DD, DD, DD, 1.f);
        dim3 g2((Mg + 127) / 128, (HH + 127) / 128);
        gemm_wmma<0, true, false, false><<<g2, 256, 0, stream>>>(
            h1c, F(P_GBFP_W2), F(P_GBFP_B2), nullptr,
            ab + (size_t)c * Mg * HH, Mg, DD, HH, DD, HH, HH, 1.f);
    }

    // ================= encoder layers =================
    const dim3 fgrid_enc(NB * HH, (SS + 31) / 32);
    for (int l = 0; l < LL; ++l) {
        const int b0 = P_ENC_BASE + l * 12;
        g_plain(x, F(b0 + 0), nullptr, nullptr, qb, (int)NS, DD, DD);
        g_plain(x, F(b0 + 1), nullptr, nullptr, kb, (int)NS, DD, DD);
        g_plain(x, F(b0 + 2), nullptr, nullptr, vb, (int)NS, DD, DD);
        flash_attn<<<fgrid_enc, 64, 0, stream>>>(
            qb, kb, vb, ob, Ii(IN_SRC_SEQ), inv, ab, SS, SS, 0);
        g_plain(ob, F(b0 + 3), nullptr, x, x, (int)NS, DD, DD);
        ln_kernel<<<(unsigned)NS, 256, 0, stream>>>(x, F(b0 + 4), F(b0 + 5));
        g_relu_bf(x, F(b0 + 6), F(b0 + 7), hffn, (int)NS, DD, DFFF);
        g_abf(hffn, F(b0 + 8), F(b0 + 9), x, x, (int)NS, DFFF, DD);
        ln_kernel<<<(unsigned)NS, 256, 0, stream>>>(x, F(b0 + 10), F(b0 + 11));
    }

    // ================= decoder =================
    embed_kernel<<<(unsigned)((NS * DD + 255) / 256), 256, 0, stream>>>(
        Ii(IN_TRG_SEQ), F(P_EMB), y, nullptr, TT);
    ln_kernel<<<(unsigned)NS, 256, 0, stream>>>(y, F(P_DEC_LN_G), F(P_DEC_LN_B));

    const dim3 fgrid_dec(NB * HH, (TT + 31) / 32);
    for (int l = 0; l < LL; ++l) {
        const int b0 = P_ENC_BASE + 72 + l * 18;
        // self attention (causal)
        g_plain(y, F(b0 + 0), nullptr, nullptr, qb, (int)NS, DD, DD);
        g_plain(y, F(b0 + 1), nullptr, nullptr, kb, (int)NS, DD, DD);
        g_plain(y, F(b0 + 2), nullptr, nullptr, vb, (int)NS, DD, DD);
        flash_attn<<<fgrid_dec, 64, 0, stream>>>(
            qb, kb, vb, ob, Ii(IN_TRG_SEQ), nullptr, nullptr, TT, TT, 1);
        g_plain(ob, F(b0 + 3), nullptr, y, y, (int)NS, DD, DD);
        ln_kernel<<<(unsigned)NS, 256, 0, stream>>>(y, F(b0 + 4), F(b0 + 5));
        // cross attention (kv from encoder output x)
        g_plain(y, F(b0 + 6), nullptr, nullptr, qb, (int)NS, DD, DD);
        g_plain(x, F(b0 + 7), nullptr, nullptr, kb, (int)NS, DD, DD);
        g_plain(x, F(b0 + 8), nullptr, nullptr, vb, (int)NS, DD, DD);
        flash_attn<<<fgrid_dec, 64, 0, stream>>>(
            qb, kb, vb, ob, Ii(IN_SRC_SEQ), nullptr, nullptr, TT, SS, 2);
        g_plain(ob, F(b0 + 9), nullptr, y, y, (int)NS, DD, DD);
        ln_kernel<<<(unsigned)NS, 256, 0, stream>>>(y, F(b0 + 10), F(b0 + 11));
        // ffn
        g_relu_bf(y, F(b0 + 12), F(b0 + 13), hffn, (int)NS, DD, DFFF);
        g_abf(hffn, F(b0 + 14), F(b0 + 15), y, y, (int)NS, DFFF, DD);
        ln_kernel<<<(unsigned)NS, 256, 0, stream>>>(y, F(b0 + 16), F(b0 + 17));
    }

    // ================= tied-embedding logits =================
    {
        dim3 g((unsigned)((NS + 127) / 128), (VV + 127) / 128);
        gemm_wmma<0, false, true, false><<<g, 256, 0, stream>>>(
            y, F(P_EMB), nullptr, nullptr, (float*)d_out,
            (int)NS, DD, VV, DD, DD, VV, 0.044194173824159216f);
    }
}
